// KANLayer_82051055223083
// MI455X (gfx1250) — matrix operand order repaired and checked by reading, not compiled, not run on hardware
//
#include <hip/hip_runtime.h>
#include <hip/hip_bf16.h>

typedef __bf16 bf16_t;
typedef __attribute__((ext_vector_type(16))) __bf16 v16bf;
typedef __attribute__((ext_vector_type(8)))  float  v8f;

#define BATCH 8192
#define IN_F  1024
#define OUT_F 1024
#define SPL_C 8                      // GRID + K = 5 + 3
#define KDIM  (IN_F * 9)             // 1024 (base) + 1024*8 (spline) = 9216
#define TILE_K 32
#define LDS_STRIDE 40                // 32 halves + 8 pad -> conflict-free b128
#define NK (KDIM / TILE_K)           // 288
#define BUFB (128 * LDS_STRIDE * 2)  // bytes per LDS stage buffer (10240)

#define A_BYTES ((size_t)BATCH * KDIM * 2)   // bf16 activation matrix

// ---------------------------------------------------------------------------
// Kernel 1: Cox-de-Boor basis (fp32, rcp-based) -> packed bf16 A [x | basis*8]
// ---------------------------------------------------------------------------
__global__ __launch_bounds__(256) void kan_basis_kernel(
    const float* __restrict__ x, const float* __restrict__ grid,
    bf16_t* __restrict__ Abuf)
{
  const int idx = blockIdx.x * 256 + threadIdx.x;   // 0 .. BATCH*IN_F-1
  const int b = idx >> 10;
  const int i = idx & 1023;
  const float xv = x[idx];

  // 12 knots, 16B-aligned rows (48 bytes each)
  const float4* g4 = (const float4*)(grid + i * 12);
  const float4 t0 = g4[0], t1 = g4[1], t2 = g4[2];
  const float t[12] = { t0.x, t0.y, t0.z, t0.w,
                        t1.x, t1.y, t1.z, t1.w,
                        t2.x, t2.y, t2.z, t2.w };

  float bb[11];
#pragma unroll
  for (int j = 0; j < 11; ++j)
    bb[j] = (xv >= t[j] && xv < t[j + 1]) ? 1.0f : 0.0f;

#pragma unroll
  for (int p = 1; p <= 3; ++p) {
#pragma unroll
    for (int j = 0; j + p < 11; ++j) {    // ascending: bb[j+1] still degree p-1
      const float lw = (xv - t[j])         * __builtin_amdgcn_rcpf(t[j + p] - t[j]);
      const float rw = (t[j + p + 1] - xv) * __builtin_amdgcn_rcpf(t[j + p + 1] - t[j + 1]);
      bb[j] = lw * bb[j] + rw * bb[j + 1];
    }
  }

  const size_t rowBase = (size_t)b * KDIM;
  Abuf[rowBase + i] = (bf16_t)xv;                 // base-GEMM column

  union { bf16_t h[8]; int4 q; } pk;
#pragma unroll
  for (int c = 0; c < 8; ++c) pk.h[c] = (bf16_t)bb[c];
  *(int4*)(Abuf + rowBase + IN_F + (size_t)i * 8) = pk.q;   // 16B aligned
}

// ---------------------------------------------------------------------------
// Kernel 2: concat + convert weights -> bf16 W (OUT_F x KDIM), 8 elems/thread
// ---------------------------------------------------------------------------
__global__ __launch_bounds__(256) void kan_wcat_kernel(
    const float* __restrict__ bw, const float* __restrict__ sw,
    bf16_t* __restrict__ Wbuf)
{
  const int idx = blockIdx.x * 256 + threadIdx.x;  // 0 .. OUT_F*KDIM/8-1
  const int o  = idx / (KDIM / 8);
  const int k  = (idx - o * (KDIM / 8)) * 8;       // 8-aligned, never straddles 1024
  float4 lo, hh;
  if (k < IN_F) {
    const float4* p = (const float4*)(bw + (size_t)o * IN_F + k);
    lo = p[0]; hh = p[1];
  } else {
    const float4* p = (const float4*)(sw + (size_t)o * (IN_F * SPL_C) + (k - IN_F));
    lo = p[0]; hh = p[1];
  }
  union { bf16_t h[8]; int4 q; } pk;
  pk.h[0] = (bf16_t)lo.x; pk.h[1] = (bf16_t)lo.y;
  pk.h[2] = (bf16_t)lo.z; pk.h[3] = (bf16_t)lo.w;
  pk.h[4] = (bf16_t)hh.x; pk.h[5] = (bf16_t)hh.y;
  pk.h[6] = (bf16_t)hh.z; pk.h[7] = (bf16_t)hh.w;
  *(int4*)(Wbuf + (size_t)o * KDIM + k) = pk.q;
}

// ---------------------------------------------------------------------------
// Kernel 3: bf16 WMMA GEMM  C[8192x1024] = A[8192x9216] * W[1024x9216]^T
//           global->LDS via gfx1250 async copies (ASYNCcnt), double-buffered
// ---------------------------------------------------------------------------
union FragU { int4 q[2]; v16bf v; };

__global__ __launch_bounds__(256) void kan_gemm_kernel(
    const bf16_t* __restrict__ A, const bf16_t* __restrict__ W,
    float* __restrict__ C)
{
  __shared__ bf16_t sA[2][128 * LDS_STRIDE];
  __shared__ bf16_t sB[2][128 * LDS_STRIDE];

  const int t  = threadIdx.x;
  const int M0 = blockIdx.y * 128;
  const int N0 = blockIdx.x * 128;
  const int w  = t >> 5;
  const int l  = t & 31;
  const int lr = l & 15;
  const int hi = l >> 4;           // which half-wave
  const int wm = w & 3;            // 4 waves along M (32 rows each)
  const int wn = w >> 2;           // 2 waves along N (64 cols each)

  // global->LDS staging: each thread async-copies 2x16B per matrix per stage
  const int rA   = t >> 2;                 // 0..63
  const int colh = (t & 3) * 8;            // half-offset within 32-wide k-tile
  const bf16_t* gA0 = A + (size_t)(M0 + rA)      * KDIM + colh;
  const bf16_t* gA1 = A + (size_t)(M0 + rA + 64) * KDIM + colh;
  const bf16_t* gB0 = W + (size_t)(N0 + rA)      * KDIM + colh;
  const bf16_t* gB1 = W + (size_t)(N0 + rA + 64) * KDIM + colh;
  const int s0 = rA * LDS_STRIDE + colh;
  const int s1 = (rA + 64) * LDS_STRIDE + colh;

  // raw LDS byte offsets for the async-copy VDST operand
  const unsigned ldsA0 = (unsigned)(uintptr_t)&sA[0][s0];
  const unsigned ldsA1 = (unsigned)(uintptr_t)&sA[0][s1];
  const unsigned ldsB0 = (unsigned)(uintptr_t)&sB[0][s0];
  const unsigned ldsB1 = (unsigned)(uintptr_t)&sB[0][s1];

#define ASYNC_STAGE(BUF, KT)                                                   \
  {                                                                            \
    const size_t ko_ = (size_t)(KT) * TILE_K;                                  \
    asm volatile("global_load_async_to_lds_b128 %0, %1, off"                   \
                 :: "v"(ldsA0 + (BUF) * BUFB), "v"((const void*)(gA0 + ko_))   \
                 : "memory");                                                  \
    asm volatile("global_load_async_to_lds_b128 %0, %1, off"                   \
                 :: "v"(ldsA1 + (BUF) * BUFB), "v"((const void*)(gA1 + ko_))   \
                 : "memory");                                                  \
    asm volatile("global_load_async_to_lds_b128 %0, %1, off"                   \
                 :: "v"(ldsB0 + (BUF) * BUFB), "v"((const void*)(gB0 + ko_))   \
                 : "memory");                                                  \
    asm volatile("global_load_async_to_lds_b128 %0, %1, off"                   \
                 :: "v"(ldsB1 + (BUF) * BUFB), "v"((const void*)(gB1 + ko_))   \
                 : "memory");                                                  \
  }

  // A frag: lane<16 -> K{0..7,16..23}; lane>=16 -> K{8..15,24..31}
  // B frag: lane<16 -> K{0..7,16..23}; lane>=16 -> K{8..15,24..31}
#define COMPUTE_STAGE(BUF)                                                     \
  {                                                                            \
    FragU fa[2], fb[4];                                                        \
    _Pragma("unroll")                                                          \
    for (int f = 0; f < 2; ++f) {                                              \
      const int row = wm * 32 + f * 16 + lr;                                   \
      fa[f].q[0] = *(const int4*)&sA[BUF][row * LDS_STRIDE + hi * 8];          \
      fa[f].q[1] = *(const int4*)&sA[BUF][row * LDS_STRIDE + 16 + hi * 8];     \
    }                                                                          \
    _Pragma("unroll")                                                          \
    for (int g = 0; g < 4; ++g) {                                              \
      const int row = wn * 64 + g * 16 + lr;                                   \
      fb[g].q[0] = *(const int4*)&sB[BUF][row * LDS_STRIDE + hi * 8];          \
      fb[g].q[1] = *(const int4*)&sB[BUF][row * LDS_STRIDE + 16 + hi * 8];     \
    }                                                                          \
    _Pragma("unroll")                                                          \
    for (int f = 0; f < 2; ++f)                                                \
      _Pragma("unroll")                                                        \
      for (int g = 0; g < 4; ++g)                                              \
        acc[f][g] = __builtin_amdgcn_wmma_f32_16x16x32_bf16(                   \
            false, fa[f].v, false, fb[g].v, (short)0, acc[f][g], false, false);\
  }

#define WAIT_ASYNC_AND_BARRIER()                                \
  {                                                             \
    asm volatile("s_wait_asynccnt 0x0" ::: "memory");           \
    __syncthreads();                                            \
  }

  v8f acc[2][4] = {};

  // prologue: stage 0 -> buf0
  ASYNC_STAGE(0, 0);
  WAIT_ASYNC_AND_BARRIER();

#pragma unroll 1
  for (int k2 = 0; k2 < NK / 2; ++k2) {
    // even stage: compute buf0, prefetch (2*k2+1) -> buf1
    ASYNC_STAGE(1, 2 * k2 + 1);
    COMPUTE_STAGE(0);
    WAIT_ASYNC_AND_BARRIER();

    // odd stage: compute buf1, prefetch (2*k2+2) -> buf0
    if (2 * k2 + 2 < NK) ASYNC_STAGE(0, 2 * k2 + 2);
    COMPUTE_STAGE(1);
    WAIT_ASYNC_AND_BARRIER();
  }

  // epilogue: C/D layout -> VGPR r holds M=r (lanes 0-15) / M=8+r (lanes 16-31)
#pragma unroll
  for (int f = 0; f < 2; ++f) {
#pragma unroll
    for (int g = 0; g < 4; ++g) {
      const int col = N0 + wn * 64 + g * 16 + lr;
#pragma unroll
      for (int r = 0; r < 8; ++r) {
        const int row = M0 + wm * 32 + f * 16 + hi * 8 + r;
        C[(size_t)row * OUT_F + col] = acc[f][g][r];
      }
    }
  }
#undef ASYNC_STAGE
#undef COMPUTE_STAGE
#undef WAIT_ASYNC_AND_BARRIER
}

// ---------------------------------------------------------------------------
extern "C" void kernel_launch(void* const* d_in, const int* in_sizes, int n_in,
                              void* d_out, int out_size, void* d_ws, size_t ws_size,
                              hipStream_t stream) {
  (void)in_sizes; (void)n_in; (void)out_size; (void)ws_size;
  const float* x    = (const float*)d_in[0];
  const float* bw   = (const float*)d_in[1];
  const float* sw   = (const float*)d_in[2];
  const float* grid = (const float*)d_in[3];
  float* out = (float*)d_out;

  bf16_t* Abuf = (bf16_t*)d_ws;                          // 151 MB
  bf16_t* Wbuf = (bf16_t*)((char*)d_ws + A_BYTES);       // +19 MB

  kan_basis_kernel<<<(BATCH * IN_F) / 256, 256, 0, stream>>>(x, grid, Abuf);
  kan_wcat_kernel<<<(OUT_F * KDIM) / (8 * 256), 256, 0, stream>>>(bw, sw, Wbuf);

  dim3 g(OUT_F / 128, BATCH / 128);                      // (8, 64)
  kan_gemm_kernel<<<g, 256, 0, stream>>>(Abuf, Wbuf, out);
}
